// AssociationLoss_14053132992797
// MI455X (gfx1250) — matrix operand order repaired and checked by reading, not compiled, not run on hardware
//
#include <hip/hip_runtime.h>
#include <stdint.h>

// ---------------------------------------------------------------------------
// AssociationLoss for MI455X (gfx1250): bf16 WMMA GEMMs with async-LDS
// double-buffered B panels + streaming epilogues.  N=2, C=512, H=W=64, L=4096
// ---------------------------------------------------------------------------

typedef __attribute__((ext_vector_type(16))) __bf16 v16bf;
typedef __attribute__((ext_vector_type(8)))  float  v8f;

#define NB    2
#define CCH   512
#define LPIX  4096
#define EPSF  1e-10f

// f32 -> bf16 (round to nearest even, inf/nan passthrough)
__device__ __forceinline__ unsigned short f2bf(float f) {
  union { float f; uint32_t u; } x; x.f = f;
  uint32_t u = x.u;
  uint32_t r;
  if ((u & 0x7F800000u) == 0x7F800000u) r = u >> 16;
  else r = (u + 0x7FFFu + ((u >> 16) & 1u)) >> 16;
  return (unsigned short)r;
}

union Frag {
  v16bf v;
  uint4 q[2];
};

// async 16B global -> LDS copy (gfx1250, tracked by ASYNCcnt)
__device__ __forceinline__ void asyncCopy16(uint32_t ldsOff, const void* g) {
  asm volatile("global_load_async_to_lds_b128 %0, %1, off"
               :: "v"(ldsOff), "v"(g) : "memory");
}
__device__ __forceinline__ void waitAsync() {
  asm volatile("s_wait_asynccnt 0" ::: "memory");
}

// ---------------------------------------------------------------------------
// per-pixel 1/max(||x||,1e-12) over channels; X is [n][c][l] channel-major
// ---------------------------------------------------------------------------
__global__ void pixel_rnorm(const float* __restrict__ X, float* __restrict__ rn) {
  int idx = blockIdx.x * blockDim.x + threadIdx.x;     // over N*L
  int n = idx >> 12, l = idx & (LPIX - 1);
  const float* Xb = X + (size_t)n * CCH * LPIX + l;
  float s = 0.f;
  for (int c = 0; c < CCH; ++c) { float v = Xb[(size_t)c * LPIX]; s += v * v; }
  rn[idx] = 1.f / fmaxf(sqrtf(s), 1e-12f);
}

// ---------------------------------------------------------------------------
// normalize + transpose: f32 [n][c][l] -> bf16 [n][l][c]  (32x32 LDS tiles)
// ---------------------------------------------------------------------------
__global__ void norm_transpose(const float* __restrict__ X, const float* __restrict__ rn,
                               unsigned short* __restrict__ Out) {
  __shared__ float tile[32][33];
  int n = blockIdx.z;
  int lBase = blockIdx.x * 32;
  int cBase = blockIdx.y * 32;
  const float* Xb = X + (size_t)n * CCH * LPIX;
  const float* rb = rn + (size_t)n * LPIX;
  int tx = threadIdx.x, ty = threadIdx.y;
  float scale = rb[lBase + tx];
#pragma unroll
  for (int k = 0; k < 4; ++k) {
    int c = cBase + ty + k * 8;
    tile[ty + k * 8][tx] = Xb[(size_t)c * LPIX + lBase + tx] * scale;
  }
  __syncthreads();
  unsigned short* Ob = Out + (size_t)n * LPIX * CCH;
#pragma unroll
  for (int k = 0; k < 4; ++k) {
    int l = lBase + ty + k * 8;
    Ob[(size_t)l * CCH + cBase + tx] = f2bf(tile[tx][ty + k * 8]);
  }
}

// ---------------------------------------------------------------------------
// elementwise f32 -> bf16 (same layout)
// ---------------------------------------------------------------------------
__global__ void conv_to_bf16(const float* __restrict__ X, unsigned short* __restrict__ Y, int n) {
  int i = blockIdx.x * blockDim.x + threadIdx.x;
  if (i < n) Y[i] = f2bf(X[i]);
}

// ---------------------------------------------------------------------------
// WMMA A fragment loader (bf16, row-major source)
// A frag (16x32): lanes 0-15 M=0..15 K {0..7,16..23}; lanes 16-31 K {8..15,24..31}
// ---------------------------------------------------------------------------
__device__ __forceinline__ v16bf loadA(const unsigned short* row, int k, int half) {
  Frag a;
  a.q[0] = *(const uint4*)(row + k + half * 8);
  a.q[1] = *(const uint4*)(row + k + 16 + half * 8);
  return a.v;
}

// ---------------------------------------------------------------------------
// Block GEMM NT body: Out[i][j] (+= epilogue) with A: MxK, B: NcxK row-major
// bf16.  Block = 256 threads (8 waves): 128 rows x 64 cols per block.
// The shared 64xK B panel is staged through LDS in K-chunks of 64 via
// async global->LDS copies, double-buffered (ASYNCcnt), so the DMA of the
// next panel overlaps the 8 WMMAs of the current chunk.  Panel rows padded
// to 72 shorts (stride 36 dwords) -> conflict-free ds_load_b128 frag reads.
// All four B frags of a k-step are loaded before the 4 back-to-back WMMAs
// (disjoint accumulators -> no RAW hazard, single ds-wait per k-step).
// ---------------------------------------------------------------------------
#define PROWS 72  // padded row stride in shorts

template <bool AGG>
__device__ __forceinline__ void gemm_body(const unsigned short* __restrict__ A,
                                          const unsigned short* __restrict__ B,
                                          const float* __restrict__ X2,
                                          float* __restrict__ Out,
                                          int M, int Nc, int K,
                                          long aStride, long bStride, long oStride,
                                          int mGroups) {
  __shared__ unsigned short panel[2][64 * PROWS];
  const int tid = threadIdx.x;
  const int lane = tid & 31, wave = tid >> 5;
  const int half = lane >> 4;
  const int l16 = lane & 15;

  const int nGroups = Nc >> 6;
  const int gpb = mGroups * nGroups;
  const int batch = blockIdx.x / gpb;
  const int r = blockIdx.x - batch * gpb;
  const int mg = r % mGroups;
  const int ng = r / mGroups;

  const unsigned short* Ab = A + (size_t)batch * aStride;
  const unsigned short* Bb = B + (size_t)batch * bStride;
  float* Ob = Out + (size_t)batch * oStride;
  const float* Xb = AGG ? (X2 + (size_t)batch * oStride) : nullptr;

  const int mBase = mg * 128 + wave * 16;
  const int nBase0 = ng << 6;

  // stage one 64x64 bf16 B chunk: 512 x 16B, 2 chunks per thread
  const int cRow0 = tid >> 3, cSeg = tid & 7;
  const unsigned short* bRowBase = Bb + (size_t)nBase0 * K;

  v8f acc[4] = {};
  const unsigned short* aRow = Ab + (size_t)(mBase + l16) * K;

  // prologue: chunk 0 -> buffer 0
  {
#pragma unroll
    for (int c = 0; c < 2; ++c) {
      int row = cRow0 + c * 32;
      uint32_t ldsOff = (uint32_t)(uintptr_t)&panel[0][row * PROWS + cSeg * 8];
      asyncCopy16(ldsOff, bRowBase + (size_t)row * K + cSeg * 8);
    }
  }

  int cur = 0;
  for (int kc = 0; kc < K; kc += 64, cur ^= 1) {
    waitAsync();          // this wave's copies into panel[cur] done
    __syncthreads();      // all waves' copies done; prev reads of panel[cur^1] done
    if (kc + 64 < K) {
#pragma unroll
      for (int c = 0; c < 2; ++c) {
        int row = cRow0 + c * 32;
        uint32_t ldsOff = (uint32_t)(uintptr_t)&panel[cur ^ 1][row * PROWS + cSeg * 8];
        asyncCopy16(ldsOff, bRowBase + (size_t)row * K + kc + 64 + cSeg * 8);
      }
    }
#pragma unroll
    for (int k2 = 0; k2 < 64; k2 += 32) {
      v16bf a = loadA(aRow, kc + k2, half);
      Frag b[4];
#pragma unroll
      for (int t = 0; t < 4; ++t) {
        const unsigned short* bp =
            &panel[cur][(t * 16 + l16) * PROWS + k2 + half * 8];
        b[t].q[0] = *(const uint4*)(bp);
        b[t].q[1] = *(const uint4*)(bp + 16);
      }
#pragma unroll
      for (int t = 0; t < 4; ++t)
        acc[t] = __builtin_amdgcn_wmma_f32_16x16x32_bf16(false, a, false, b[t].v,
                                                         (short)0, acc[t], false, false);
    }
  }

#pragma unroll
  for (int t = 0; t < 4; ++t) {
    int col = nBase0 + t * 16 + l16;
#pragma unroll
    for (int rr = 0; rr < 8; ++rr) {
      size_t o = (size_t)(mBase + rr + half * 8) * Nc + col;
      if (AGG) Ob[o] = 0.5f * Xb[o] + 0.5f * acc[t][rr];
      else     Ob[o] = acc[t][rr];
    }
  }
}

__global__ void gemm_nt_bf16(const unsigned short* __restrict__ A,
                             const unsigned short* __restrict__ B,
                             float* __restrict__ Out,
                             int M, int Nc, int K,
                             long aStride, long bStride, long oStride,
                             int mGroups) {
  gemm_body<false>(A, B, nullptr, Out, M, Nc, K, aStride, bStride, oStride, mGroups);
}

__global__ void gemm_agg_bf16(const unsigned short* __restrict__ A,
                              const unsigned short* __restrict__ B,
                              const float* __restrict__ X2,
                              float* __restrict__ Out,
                              int M, int Nc, int K,
                              long aStride, long bStride, long oStride,
                              int mGroups) {
  gemm_body<true>(A, B, X2, Out, M, Nc, K, aStride, bStride, oStride, mGroups);
}

// ---------------------------------------------------------------------------
// standardized row softmax -> bf16 (for sim_self scoring). One block per row.
// ---------------------------------------------------------------------------
__global__ void score_rows_bf16(const float* __restrict__ S, unsigned short* __restrict__ Out) {
  __shared__ float row[LPIX];
  __shared__ float red[256];
  const int tid = threadIdx.x;
  const float* src = S + (size_t)blockIdx.x * LPIX;
  float s = 0.f, s2 = 0.f, mx = -3.402823e38f;
  for (int j = tid; j < LPIX; j += 256) {
    float v = src[j];
    row[j] = v; s += v; s2 += v * v; mx = fmaxf(mx, v);
  }
  red[tid] = s; __syncthreads();
  for (int o = 128; o > 0; o >>= 1) { if (tid < o) red[tid] += red[tid + o]; __syncthreads(); }
  float sumAll = red[0]; __syncthreads();
  red[tid] = s2; __syncthreads();
  for (int o = 128; o > 0; o >>= 1) { if (tid < o) red[tid] += red[tid + o]; __syncthreads(); }
  float sum2All = red[0]; __syncthreads();
  red[tid] = mx; __syncthreads();
  for (int o = 128; o > 0; o >>= 1) { if (tid < o) red[tid] = fmaxf(red[tid], red[tid + o]); __syncthreads(); }
  float mxAll = red[0]; __syncthreads();

  const float mean = sumAll * (1.f / LPIX);
  const float var = (sum2All - (float)LPIX * mean * mean) / (float)(LPIX - 1);
  const float inv = 1.f / (sqrtf(fmaxf(var, 0.f)) + EPSF);
  const float zmax = (mxAll - mean) * inv;

  float se = 0.f;
  for (int j = tid; j < LPIX; j += 256) se += __expf((row[j] - mean) * inv - zmax);
  red[tid] = se; __syncthreads();
  for (int o = 128; o > 0; o >>= 1) { if (tid < o) red[tid] += red[tid + o]; __syncthreads(); }
  const float rs = 1.f / red[0]; __syncthreads();

  unsigned short* dst = Out + (size_t)blockIdx.x * LPIX;
  for (int j = tid; j < LPIX; j += 256)
    dst[j] = f2bf(__expf((row[j] - mean) * inv - zmax) * rs);
}

// ---------------------------------------------------------------------------
// per-row: argmax (first max) + softmax value at argmax = 1/sum(exp(z-zmax))
// ---------------------------------------------------------------------------
__global__ void row_assoc(const float* __restrict__ S,
                          float* __restrict__ assocSim, int* __restrict__ midIdx) {
  __shared__ float row[LPIX];
  __shared__ float red[256];
  __shared__ float rv[256];
  __shared__ int   ri[256];
  const int tid = threadIdx.x;
  const float* src = S + (size_t)blockIdx.x * LPIX;
  float s = 0.f, s2 = 0.f, mx = -3.402823e38f; int bi = 0;
  for (int j = tid; j < LPIX; j += 256) {
    float v = src[j];
    row[j] = v; s += v; s2 += v * v;
    if (v > mx) { mx = v; bi = j; }
  }
  red[tid] = s; __syncthreads();
  for (int o = 128; o > 0; o >>= 1) { if (tid < o) red[tid] += red[tid + o]; __syncthreads(); }
  float sumAll = red[0]; __syncthreads();
  red[tid] = s2; __syncthreads();
  for (int o = 128; o > 0; o >>= 1) { if (tid < o) red[tid] += red[tid + o]; __syncthreads(); }
  float sum2All = red[0]; __syncthreads();
  rv[tid] = mx; ri[tid] = bi; __syncthreads();
  for (int o = 128; o > 0; o >>= 1) {
    if (tid < o) {
      float ov = rv[tid + o]; int oi = ri[tid + o];
      if (ov > rv[tid] || (ov == rv[tid] && oi < ri[tid])) { rv[tid] = ov; ri[tid] = oi; }
    }
    __syncthreads();
  }
  float mxAll = rv[0]; int bIdx = ri[0]; __syncthreads();

  const float mean = sumAll * (1.f / LPIX);
  const float var = (sum2All - (float)LPIX * mean * mean) / (float)(LPIX - 1);
  const float inv = 1.f / (sqrtf(fmaxf(var, 0.f)) + EPSF);
  const float zmax = (mxAll - mean) * inv;

  float se = 0.f;
  for (int j = tid; j < LPIX; j += 256) se += __expf((row[j] - mean) * inv - zmax);
  red[tid] = se; __syncthreads();
  for (int o = 128; o > 0; o >>= 1) { if (tid < o) red[tid] += red[tid + o]; __syncthreads(); }
  if (tid == 0) {
    assocSim[blockIdx.x] = 1.f / red[0];
    midIdx[blockIdx.x] = bIdx;
  }
}

// ---------------------------------------------------------------------------
// per-column stats: one thread per column, coalesced row sweeps
// ---------------------------------------------------------------------------
__global__ void col_assoc(const float* __restrict__ S,
                          float* __restrict__ maxSim, int* __restrict__ maxIdx) {
  const int n = blockIdx.x >> 4;                        // 16 blocks of 256 cols per batch
  const int j = ((blockIdx.x & 15) << 8) + threadIdx.x;
  const float* base = S + (size_t)n * LPIX * LPIX;
  float s = 0.f, s2 = 0.f, mx = -3.402823e38f; int bi = 0;
  for (int i = 0; i < LPIX; ++i) {
    float v = base[(size_t)i * LPIX + j];
    s += v; s2 += v * v;
    if (v > mx) { mx = v; bi = i; }
  }
  const float mean = s * (1.f / LPIX);
  const float var = (s2 - (float)LPIX * mean * mean) / (float)(LPIX - 1);
  const float inv = 1.f / (sqrtf(fmaxf(var, 0.f)) + EPSF);
  const float zmax = (mx - mean) * inv;
  float se = 0.f;
  for (int i = 0; i < LPIX; ++i) {
    float v = base[(size_t)i * LPIX + j];
    se += __expf((v - mean) * inv - zmax);
  }
  int o = n * LPIX + j;
  maxSim[o] = 1.f / se;
  maxIdx[o] = bi;
}

// ---------------------------------------------------------------------------
// cycle association + masked BCE reduction
// ---------------------------------------------------------------------------
__global__ void init_acc(float* acc) { if (threadIdx.x < 2) acc[threadIdx.x] = 0.f; }

__global__ void bce_reduce(const int* __restrict__ gt,
                           const float* __restrict__ assocSim, const int* __restrict__ midIdx,
                           const float* __restrict__ maxSim, const int* __restrict__ maxIdx,
                           float* __restrict__ acc) {
  __shared__ float rs[256], rc[256];
  const int tid = threadIdx.x;
  const int idx = blockIdx.x * 256 + tid;               // over N*L
  const int n = idx >> 12;
  const int base = n * LPIX;
  const int g = gt[idx];
  const int mid = midIdx[idx];
  const int idx2 = maxIdx[base + mid];
  const float sim = assocSim[idx] * maxSim[base + mid];
  const int g2 = gt[base + idx2];
  const bool valid = (g != 255);
  const bool sel = (g == g2) && valid;
  float lsum = 0.f, lcnt = 0.f;
  if (sel) {
    float lg = fmaxf(__logf(fmaxf(sim, 0.f)), -100.f);
    lsum = -lg; lcnt = 1.f;
  }
  rs[tid] = lsum; rc[tid] = lcnt; __syncthreads();
  for (int o = 128; o > 0; o >>= 1) {
    if (tid < o) { rs[tid] += rs[tid + o]; rc[tid] += rc[tid + o]; }
    __syncthreads();
  }
  if (tid == 0) {
    atomicAdd(&acc[0], rs[0]);
    atomicAdd(&acc[1], rc[0]);
  }
}

__global__ void finalize(const float* __restrict__ acc, float* __restrict__ out) {
  float s = acc[0], c = acc[1];
  out[0] = (c > 0.f) ? s / fmaxf(c, 1.f) : 0.f;
}

// ---------------------------------------------------------------------------
// launch
// ---------------------------------------------------------------------------
extern "C" void kernel_launch(void* const* d_in, const int* in_sizes, int n_in,
                              void* d_out, int out_size, void* d_ws, size_t ws_size,
                              hipStream_t stream) {
  (void)in_sizes; (void)n_in; (void)out_size; (void)ws_size;
  const float* x1 = (const float*)d_in[0];
  const float* x2 = (const float*)d_in[1];
  const int*   gt = (const int*)d_in[2];
  char* ws = (char*)d_ws;

  // workspace layout (bytes)
  const size_t OFF_RAW   = 0;                              // f32  [N,L,L]  128MB (reused)
  const size_t OFF_SIMB  = OFF_RAW   + (size_t)NB*LPIX*LPIX*4;   // bf16 [N,L,L]  64MB
  const size_t OFF_X2N   = OFF_SIMB  + (size_t)NB*LPIX*LPIX*2;   // bf16 [N,L,C]
  const size_t OFF_X1N   = OFF_X2N   + (size_t)NB*LPIX*CCH*2;
  const size_t OFF_X2CM  = OFF_X1N   + (size_t)NB*LPIX*CCH*2;    // bf16 [N,C,L]
  const size_t OFF_X2NEW = OFF_X2CM  + (size_t)NB*CCH*LPIX*2;    // f32  [N,C,L]
  const size_t OFF_X2NN  = OFF_X2NEW + (size_t)NB*CCH*LPIX*4;    // bf16 [N,L,C]
  const size_t OFF_RN    = OFF_X2NN  + (size_t)NB*LPIX*CCH*2;    // f32  [N,L]
  const size_t OFF_AS    = OFF_RN    + (size_t)NB*LPIX*4;        // assoc_sim
  const size_t OFF_MI    = OFF_AS    + (size_t)NB*LPIX*4;        // mid_idx
  const size_t OFF_MS    = OFF_MI    + (size_t)NB*LPIX*4;        // max_sim
  const size_t OFF_MX    = OFF_MS    + (size_t)NB*LPIX*4;        // max_idx
  const size_t OFF_ACC   = OFF_MX    + (size_t)NB*LPIX*4;

  float*          RAW   = (float*)(ws + OFF_RAW);
  unsigned short* SIMB  = (unsigned short*)(ws + OFF_SIMB);
  unsigned short* X2N   = (unsigned short*)(ws + OFF_X2N);
  unsigned short* X1N   = (unsigned short*)(ws + OFF_X1N);
  unsigned short* X2CM  = (unsigned short*)(ws + OFF_X2CM);
  float*          X2NEW = (float*)(ws + OFF_X2NEW);
  unsigned short* X2NN  = (unsigned short*)(ws + OFF_X2NN);
  float*          RN    = (float*)(ws + OFF_RN);
  float*          AS    = (float*)(ws + OFF_AS);
  int*            MI    = (int*)(ws + OFF_MI);
  float*          MS    = (float*)(ws + OFF_MS);
  int*            MX    = (int*)(ws + OFF_MX);
  float*          ACC   = (float*)(ws + OFF_ACC);

  const long LC = (long)LPIX * CCH;
  const long LL = (long)LPIX * LPIX;
  const dim3 tb(32, 8);
  const dim3 tg(LPIX / 32, CCH / 32, NB);

  // --- normalize x2 (original) -> pixel-major bf16 ---
  pixel_rnorm<<<NB * LPIX / 256, 256, 0, stream>>>(x2, RN);
  norm_transpose<<<tg, tb, 0, stream>>>(x2, RN, X2N);
  // --- normalize x1 -> pixel-major bf16 ---
  pixel_rnorm<<<NB * LPIX / 256, 256, 0, stream>>>(x1, RN);
  norm_transpose<<<tg, tb, 0, stream>>>(x1, RN, X1N);
  // --- x2 channel-major bf16 (A operand of agg GEMM) ---
  conv_to_bf16<<<(NB * CCH * LPIX) / 256, 256, 0, stream>>>(x2, X2CM, NB * CCH * LPIX);

  // --- sim_self = X2n . X2n^T  ->  RAW (f32) ---
  gemm_nt_bf16<<<NB * (LPIX / 128) * (LPIX / 64), 256, 0, stream>>>(
      X2N, X2N, RAW, LPIX, LPIX, CCH, LC, LC, LL, LPIX / 128);
  // --- scored sim_self -> bf16 ---
  score_rows_bf16<<<NB * LPIX, 256, 0, stream>>>(RAW, SIMB);
  // --- x2_new = 0.5*x2 + 0.5 * X2 . sim^T ---
  gemm_agg_bf16<<<NB * (CCH / 128) * (LPIX / 64), 256, 0, stream>>>(
      X2CM, SIMB, x2, X2NEW, CCH, LPIX, LPIX, LC, LL, LC, CCH / 128);
  // --- normalize x2_new -> pixel-major bf16 ---
  pixel_rnorm<<<NB * LPIX / 256, 256, 0, stream>>>(X2NEW, RN);
  norm_transpose<<<tg, tb, 0, stream>>>(X2NEW, RN, X2NN);

  // --- raw12 = X1n . X2nn^T -> RAW (reused) ---
  gemm_nt_bf16<<<NB * (LPIX / 128) * (LPIX / 64), 256, 0, stream>>>(
      X1N, X2NN, RAW, LPIX, LPIX, CCH, LC, LC, LL, LPIX / 128);

  // --- cycle association stats ---
  row_assoc<<<NB * LPIX, 256, 0, stream>>>(RAW, AS, MI);
  col_assoc<<<NB * 16, 256, 0, stream>>>(RAW, MS, MX);

  // --- masked BCE ---
  init_acc<<<1, 64, 0, stream>>>(ACC);
  bce_reduce<<<NB * LPIX / 256, 256, 0, stream>>>(gt, AS, MI, MS, MX, ACC);
  finalize<<<1, 1, 0, stream>>>(ACC, (float*)d_out);
}
